// DKTAccumModel_55284819034428
// MI455X (gfx1250) — matrix operand order repaired and checked by reading, not compiled, not run on hardware
//
#include <hip/hip_runtime.h>
#include <stdint.h>

// ---------------- problem constants ----------------
#define BB   64
#define TT   1024
#define KKq  256      // K (output classes / q dim)
#define TWOK 512
#define EE   128
#define HH   256
#define ROWS (BB*TT)  // 65536

// ---------------- CDNA5 WMMA types ----------------
typedef __attribute__((ext_vector_type(16))) __bf16       v16bf;
typedef __attribute__((ext_vector_type(8)))  float        v8f;
typedef __attribute__((ext_vector_type(4)))  unsigned int u32x4;

union Frag { v16bf v; u32x4 q[2]; };

__device__ __forceinline__ uint32_t f2bf(float x) {
  uint32_t u = __float_as_uint(x);
  return (u + 0x7FFFu + ((u >> 16) & 1u)) >> 16;   // RNE f32 -> bf16
}
__device__ __forceinline__ float sigmoidf_(float x) { return 1.0f / (1.0f + __expf(-x)); }

// async global->LDS b128 copy (CDNA5): vdst = LDS byte address, vaddr = 64-bit global addr.
// Generic pointers to LDS carry the LDS offset in their low 32 bits (flat aperture rule).
__device__ __forceinline__ void async_g2l_b128(const void* lds_generic, const void* gaddr) {
  uint32_t l = (uint32_t)(uintptr_t)lds_generic;
  asm volatile("global_load_async_to_lds_b128 %0, %1, off" :: "v"(l), "v"(gaddr) : "memory");
}
__device__ __forceinline__ void async_wait0() {
  asm volatile("s_wait_asynccnt 0x0" ::: "memory");
}

// ---------------- barrier init ----------------
__global__ void init_bar(unsigned* bar) {
  if (threadIdx.x < 2) bar[threadIdx.x] = 0u;
}

// ---------------- per-(b,t): s = sum_k x, ed = exp(-(delta*Wd+bd)) ----------------
__global__ __launch_bounds__(256) void rowsum_ed(const float* __restrict__ x,
                                                 const float* __restrict__ delta,
                                                 const float* __restrict__ Wd,
                                                 const float* __restrict__ bd,
                                                 float* __restrict__ s,
                                                 float* __restrict__ ed) {
  int row  = blockIdx.x * 8 + (threadIdx.x >> 5);
  int lane = threadIdx.x & 31;
  const float* xr = x + (size_t)row * TWOK;
  float acc = 0.f;
  for (int i = lane; i < TWOK; i += 32) acc += xr[i];
  for (int off = 16; off; off >>= 1) acc += __shfl_down(acc, off, 32);
  if (lane == 0) {
    s[row]  = acc;
    ed[row] = __expf(-(delta[row] * Wd[0] + bd[0]));
  }
}

// ---------------- count recurrence c_t = s_t*c_{t-1} + x_t ; clog = bf16(log1p(c)) ----------------
__global__ __launch_bounds__(512) void count_scan(const float* __restrict__ x,
                                                  const float* __restrict__ s,
                                                  uint16_t* __restrict__ clog) {
  int b = blockIdx.x;       // 0..63
  int k = threadIdx.x;      // 0..511
  float c = 0.f;
  size_t base = (size_t)b * TT * TWOK;
  const float* sb = s + (size_t)b * TT;
  for (int t = 0; t < TT; ++t) {
    c = sb[t] * c + x[base + (size_t)t * TWOK + k];
    clog[base + (size_t)t * TWOK + k] = (uint16_t)f2bf(log1pf(c));
  }
}

// ---------------- pack f32 weight [K,N] into WMMA bf16 B-fragment order ----------------
// out dword index = ((ntile*(K/32)+kc)*32 + lane)*8 + v ; lane: n=lane&15, kh=lane>>4 ;
// dword v holds K = kc*32 + (v>>2)*16 + kh*8 + 2*(v&3) (+1 in high half)
__global__ __launch_bounds__(256) void pack_b(const float* __restrict__ W, int K, int N,
                                              uint32_t* __restrict__ out) {
  int id   = blockIdx.x * 256 + threadIdx.x;
  int v    = id & 7;
  int lane = (id >> 3) & 31;
  int rest = id >> 8;
  int KC   = K >> 5;
  int kc   = rest % KC;
  int nt   = rest / KC;
  int n = nt * 16 + (lane & 15);
  int k = kc * 32 + (v >> 2) * 16 + (lane >> 4) * 8 + (v & 3) * 2;
  uint32_t lo = f2bf(W[(size_t)k * N + n]);
  uint32_t hi = f2bf(W[(size_t)(k + 1) * N + n]);
  out[id] = lo | (hi << 16);
}

// ---------------- generic bf16 WMMA GEMM: C[65536 x N] = A @ Bpacked (+epilogue) ----------------
// block = 128 threads (4 waves), tile = 64 rows x 64 cols; wave owns one 16-wide N-tile and
// 4 M-subtiles sharing each B fragment (4x B reuse). LDS staging is ping-pong double buffered;
// bf16 A inputs are staged with async global->LDS b128 copies (ASYNCcnt), f32 A inputs are
// converted to bf16 through VALU on the way into LDS.
// MODE 0: bf16 store (emb, + bias)            MODE 1: f32 store + rowscale*rowvec + bias (pre)
// MODE 2: f32 store sigmoid(acc+bias)*qmask
template <int MODE, typename TA>
__global__ __launch_bounds__(128) void gemm_wmma(
    const TA* __restrict__ A, int lda,
    const uint32_t* __restrict__ Bp, int K,
    float* __restrict__ Cf, uint16_t* __restrict__ Cb, int ldc, int ncol0,
    const float* __restrict__ bias,
    const float* __restrict__ rowscale, const float* __restrict__ rowvec,
    const float* __restrict__ qmask) {
  __shared__ uint16_t sm[2][64 * 32];
  const int tid  = threadIdx.x;
  const int wave = tid >> 5;
  const int lane = tid & 31;
  const int g2   = lane >> 4;
  const int l15  = lane & 15;
  const int rowBase = blockIdx.x * 64;
  const int KC = K >> 5;

  auto stage = [&](int kc, int buf) {
    if constexpr (sizeof(TA) == 2) {
      // async copy: 2048 bf16 = 256 b128; 2 per thread
      #pragma unroll
      for (int i = 0; i < 2; ++i) {
        int e = (tid + 128 * i) * 8;
        int rr = e >> 5, kk = e & 31;
        async_g2l_b128(&sm[buf][rr * 32 + kk],
                       A + (size_t)(rowBase + rr) * lda + kc * 32 + kk);
      }
    } else {
      #pragma unroll
      for (int jq = 0; jq < 4; ++jq) {
        int idx = tid * 4 + jq * 512;
        int rr = idx >> 5, kk = idx & 31;
        size_t off = (size_t)(rowBase + rr) * lda + kc * 32 + kk;
        const float4 f = *(const float4*)(A + off);
        uint64_t pk = (uint64_t)f2bf(f.x) | ((uint64_t)f2bf(f.y) << 16) |
                      ((uint64_t)f2bf(f.z) << 32) | ((uint64_t)f2bf(f.w) << 48);
        *(uint64_t*)&sm[buf][rr * 32 + kk] = pk;
      }
    }
  };

  v8f acc[4] = {};
  stage(0, 0);
  if constexpr (sizeof(TA) == 2) async_wait0();
  __syncthreads();
  int buf = 0;
  for (int kc = 0; kc < KC; ++kc) {
    if (kc + 1 < KC) stage(kc + 1, buf ^ 1);   // overlap with this chunk's WMMAs
    Frag b;
    {
      int ntile = blockIdx.y * 4 + wave;
      size_t boff = ((size_t)(ntile * KC + kc) * 32 + lane) * 8;
      b.q[0] = *(const u32x4*)(Bp + boff);
      b.q[1] = *(const u32x4*)(Bp + boff + 4);
    }
    #pragma unroll
    for (int mt = 0; mt < 4; ++mt) {
      Frag a;
      const uint16_t* ab = &sm[buf][(mt * 16 + l15) * 32 + g2 * 8];
      a.q[0] = *(const u32x4*)ab;
      a.q[1] = *(const u32x4*)(ab + 16);
      acc[mt] = __builtin_amdgcn_wmma_f32_16x16x32_bf16(false, a.v, false, b.v, (short)0,
                                                        acc[mt], false, false);
    }
    if constexpr (sizeof(TA) == 2) async_wait0();
    __syncthreads();
    buf ^= 1;
  }
  // epilogue: C layout: lane n = l15, vgpr r -> row r + 8*g2
  int col_local = blockIdx.y * 64 + wave * 16 + l15;
  int col = ncol0 + col_local;
  #pragma unroll
  for (int mt = 0; mt < 4; ++mt) {
    #pragma unroll
    for (int r = 0; r < 8; ++r) {
      int row = rowBase + mt * 16 + r + 8 * g2;
      float vacc = acc[mt][r];
      if constexpr (MODE == 0) {
        Cb[(size_t)row * ldc + col] = (uint16_t)f2bf(vacc + bias[col_local]);
      } else if constexpr (MODE == 1) {
        Cf[(size_t)row * ldc + col] = vacc + rowscale[row] * rowvec[col] + bias[col];
      } else {
        Cf[(size_t)row * ldc + col] =
            sigmoidf_(vacc + bias[col]) * qmask[(size_t)row * ldc + col];
      }
    }
  }
}

// ---------------- persistent LSTM: 16 blocks = (4 batch tiles) x (4 H-column groups) ----------------
// Wave jsub (0..3) owns j-cols [wgj*64+jsub*16 .. +16) for ALL 4 gates -> gate math is
// register-resident (acc i/f/g/o share the same C layout). Cell state c stays in VGPRs
// across all 1024 steps. h exchanged via double-buffered bf16 global buffer (L2-resident),
// staged into LDS with async b128 copies, grid barrier between steps.
__global__ __launch_bounds__(128) void lstm_persistent(
    const float* __restrict__ pre, const uint32_t* __restrict__ Ulp,
    uint16_t* __restrict__ hex, float* __restrict__ h_all, unsigned* __restrict__ bar) {
  __shared__ uint16_t hlds[16 * HH];  // 16 batch rows x 256 h (bf16) = 8KB
  const int tid  = threadIdx.x;
  const int wave = tid >> 5;      // jsub
  const int lane = tid & 31;
  const int g2   = lane >> 4;
  const int l15  = lane & 15;
  const int btile = blockIdx.x >> 2;
  const int wgj   = blockIdx.x & 3;
  const int jcol  = wgj * 64 + wave * 16 + l15;  // H index in C layout
  const int KC = HH / 32;  // 8
  float cst[8];
  #pragma unroll
  for (int r = 0; r < 8; ++r) cst[r] = 0.f;

  for (int t = 0; t < TT; ++t) {
    if (t > 0) {
      __syncthreads();  // previous step finished reading hlds
      const uint16_t* src = hex + (size_t)((t - 1) & 1) * BB * HH + (size_t)btile * 16 * HH;
      #pragma unroll
      for (int i = 0; i < 4; ++i) {
        int e = (tid + 128 * i) * 8;           // 8 bf16 per b128
        async_g2l_b128(&hlds[e], src + e);
      }
      async_wait0();
      __syncthreads();
    }
    v8f acc[4];
    #pragma unroll
    for (int g = 0; g < 4; ++g) {
      int col = g * HH + jcol;
      #pragma unroll
      for (int r = 0; r < 8; ++r) {
        int b = btile * 16 + r + 8 * g2;
        acc[g][r] = pre[((size_t)b * TT + t) * 1024 + col];
      }
    }
    if (t > 0) {
      for (int kc = 0; kc < KC; ++kc) {
        Frag a;
        a.q[0] = *(const u32x4*)&hlds[l15 * HH + kc * 32 + g2 * 8];
        a.q[1] = *(const u32x4*)&hlds[l15 * HH + kc * 32 + 16 + g2 * 8];
        #pragma unroll
        for (int g = 0; g < 4; ++g) {
          int ntile = g * 16 + wgj * 4 + wave;
          Frag b;
          size_t boff = ((size_t)(ntile * KC + kc) * 32 + lane) * 8;
          b.q[0] = *(const u32x4*)(Ulp + boff);
          b.q[1] = *(const u32x4*)(Ulp + boff + 4);
          acc[g] = __builtin_amdgcn_wmma_f32_16x16x32_bf16(false, a.v, false, b.v, (short)0,
                                                           acc[g], false, false);
        }
      }
    }
    #pragma unroll
    for (int r = 0; r < 8; ++r) {
      float ig = sigmoidf_(acc[0][r]);
      float fg = sigmoidf_(acc[1][r]);
      float gg = tanhf(acc[2][r]);
      float og = sigmoidf_(acc[3][r]);
      float c  = fg * cst[r] + ig * gg;
      cst[r] = c;
      float h = og * tanhf(c);
      int b = btile * 16 + r + 8 * g2;
      h_all[((size_t)b * TT + t) * HH + jcol] = h;
      hex[(size_t)(t & 1) * BB * HH + (size_t)b * HH + jcol] = (uint16_t)f2bf(h);
    }
    // -------- device-wide sense-reversing barrier across 16 resident blocks --------
    __syncthreads();
    if (tid == 0) {
      __threadfence();
      unsigned gen0 = __hip_atomic_load(bar + 1, __ATOMIC_RELAXED, __HIP_MEMORY_SCOPE_AGENT);
      unsigned arr  = __hip_atomic_fetch_add(bar, 1u, __ATOMIC_ACQ_REL, __HIP_MEMORY_SCOPE_AGENT);
      if (arr == 15u) {
        __hip_atomic_store(bar, 0u, __ATOMIC_RELAXED, __HIP_MEMORY_SCOPE_AGENT);
        __hip_atomic_fetch_add(bar + 1, 1u, __ATOMIC_RELEASE, __HIP_MEMORY_SCOPE_AGENT);
      } else {
        while (__hip_atomic_load(bar + 1, __ATOMIC_ACQUIRE, __HIP_MEMORY_SCOPE_AGENT) == gen0)
          __builtin_amdgcn_s_sleep(2);
      }
    }
    __syncthreads();
  }
}

// ---------------- final row reduction: out[row] = sum_n sigbuf[row,n] ----------------
__global__ __launch_bounds__(256) void rowreduce(const float* __restrict__ buf,
                                                 float* __restrict__ out) {
  int row  = blockIdx.x * 8 + (threadIdx.x >> 5);
  int lane = threadIdx.x & 31;
  const float* r = buf + (size_t)row * KKq;
  float a = 0.f;
  for (int i = lane; i < KKq; i += 32) a += r[i];
  for (int off = 16; off; off >>= 1) a += __shfl_down(a, off, 32);
  if (lane == 0) out[row] = a;
}

// ---------------- host launcher ----------------
extern "C" void kernel_launch(void* const* d_in, const int* in_sizes, int n_in,
                              void* d_out, int out_size, void* d_ws, size_t ws_size,
                              hipStream_t stream) {
  (void)in_sizes; (void)n_in; (void)out_size; (void)ws_size;
  const float* x     = (const float*)d_in[0];
  const float* delta = (const float*)d_in[1];
  const float* q     = (const float*)d_in[2];
  const float* Wx    = (const float*)d_in[3];
  const float* bx    = (const float*)d_in[4];
  const float* Wc    = (const float*)d_in[5];
  const float* bc    = (const float*)d_in[6];
  const float* Wd    = (const float*)d_in[7];
  const float* bd    = (const float*)d_in[8];
  const float* Wl    = (const float*)d_in[9];
  const float* Ul    = (const float*)d_in[10];
  const float* bl    = (const float*)d_in[11];
  const float* Wo    = (const float*)d_in[12];
  const float* bo    = (const float*)d_in[13];
  float* out = (float*)d_out;

  char* ws = (char*)d_ws;
  size_t off = 0;
  auto alloc = [&](size_t bytes) {
    void* p = ws + off;
    off += (bytes + 255) & ~(size_t)255;
    return p;
  };
  uint16_t* clog = (uint16_t*)alloc((size_t)ROWS * TWOK * 2);  // reused later as sigbuf f32
  uint16_t* emb  = (uint16_t*)alloc((size_t)ROWS * 256 * 2);
  float*    pre  = (float*)alloc((size_t)ROWS * 1024 * 4);
  float*    hall = (float*)alloc((size_t)ROWS * HH * 4);
  float*    s    = (float*)alloc((size_t)ROWS * 4);
  float*    ed   = (float*)alloc((size_t)ROWS * 4);
  uint32_t* pWx  = (uint32_t*)alloc((size_t)TWOK * EE * 2);
  uint32_t* pWc  = (uint32_t*)alloc((size_t)TWOK * EE * 2);
  uint32_t* pWl  = (uint32_t*)alloc((size_t)256 * 1024 * 2);
  uint32_t* pUl  = (uint32_t*)alloc((size_t)HH * 1024 * 2);
  uint32_t* pWo  = (uint32_t*)alloc((size_t)HH * KKq * 2);
  uint16_t* hex  = (uint16_t*)alloc((size_t)2 * BB * HH * 2);
  unsigned* bar  = (unsigned*)alloc(256);
  float* sigbuf = (float*)clog;  // same byte size: ROWS*512*2 == ROWS*256*4

  init_bar<<<1, 64, 0, stream>>>(bar);
  rowsum_ed<<<ROWS / 8, 256, 0, stream>>>(x, delta, Wd, bd, s, ed);
  count_scan<<<BB, TWOK, 0, stream>>>(x, s, clog);
  pack_b<<<(TWOK * EE / 2) / 256, 256, 0, stream>>>(Wx, TWOK, EE, pWx);
  pack_b<<<(TWOK * EE / 2) / 256, 256, 0, stream>>>(Wc, TWOK, EE, pWc);
  pack_b<<<(256 * 1024 / 2) / 256, 256, 0, stream>>>(Wl, 256, 1024, pWl);
  pack_b<<<(HH * 1024 / 2) / 256, 256, 0, stream>>>(Ul, HH, 1024, pUl);
  pack_b<<<(HH * KKq / 2) / 256, 256, 0, stream>>>(Wo, HH, KKq, pWo);
  // emb[:,0:128] = x @ Wx + bx
  gemm_wmma<0, float><<<dim3(ROWS / 64, EE / 64), 128, 0, stream>>>(
      x, TWOK, pWx, TWOK, nullptr, emb, 256, 0, bx, nullptr, nullptr, nullptr);
  // emb[:,128:256] = log1p(count) @ Wc + bc
  gemm_wmma<0, uint16_t><<<dim3(ROWS / 64, EE / 64), 128, 0, stream>>>(
      clog, TWOK, pWc, TWOK, nullptr, emb, 256, EE, bc, nullptr, nullptr, nullptr);
  // pre = emb @ Wl[0:256] + exp_delta (x) Wl[256] + bl
  gemm_wmma<1, uint16_t><<<dim3(ROWS / 64, 1024 / 64), 128, 0, stream>>>(
      emb, 256, pWl, 256, pre, nullptr, 1024, 0, bl, ed, Wl + 256 * 1024, nullptr);
  // sequential LSTM (persistent, 16 blocks, grid barrier per step)
  lstm_persistent<<<16, 128, 0, stream>>>(pre, pUl, hex, hall, bar);
  // sigbuf = sigmoid(h @ Wo + bo) * q
  gemm_wmma<2, float><<<dim3(ROWS / 64, KKq / 64), 128, 0, stream>>>(
      hall, HH, pWo, HH, sigbuf, nullptr, KKq, 0, bo, nullptr, nullptr, q);
  // out = row-sum
  rowreduce<<<ROWS / 8, 256, 0, stream>>>(sigbuf, out);
}
